// MixGCNLayers_61083024883813
// MI455X (gfx1250) — matrix operand order repaired and checked by reading, not compiled, not run on hardware
//
#include <hip/hip_runtime.h>
#include <math.h>

// ---------------------------------------------------------------------------
// MixGCNLayers for MI455X (gfx1250, wave32).
// Dense 64x64 GEMMs -> v_wmma_f32_16x16x32_f16 (one wave = 16x64 strip),
//   A loaded as b128 f32 + cvt_pk, W pre-packed once into f16 WMMA-B layout.
// Edge aggregation -> edge-parallel f32 atomics (resolve in 192MB L2).
// ---------------------------------------------------------------------------

typedef _Float16 v16h __attribute__((ext_vector_type(16)));
typedef float    v8f  __attribute__((ext_vector_type(8)));

__device__ __forceinline__ void atomAddF(float* p, float v) {
  __hip_atomic_fetch_add(p, v, __ATOMIC_RELAXED, __HIP_MEMORY_SCOPE_AGENT);
}
__device__ __forceinline__ float lrelu(float x) { return x > 0.f ? x : 0.2f * x; }
// order-preserving float<->int key for atomicMax on signed int
__device__ __forceinline__ int   okey(float f) { int i = __float_as_int(f); return i >= 0 ? i : (i ^ 0x7fffffff); }
__device__ __forceinline__ float dekey(int k)  { return __int_as_float(k >= 0 ? k : (k ^ 0x7fffffff)); }

// ---------------------------------------------------------------------------
// Pre-pack one 64x64 f32 weight matrix into f16 WMMA B-fragment layout:
//   Wh[((ks*4+cb)*32 + lane)*16 + j] = W[(ks*32 + 8*(lane>>4) + (j&7) + 16*(j>>3))*64 + cb*16 + (lane&15)]
// so a wave loads each B fragment as one contiguous v16h (2x b128).
// ---------------------------------------------------------------------------
__global__ void pack_w16(const float* __restrict__ W, _Float16* __restrict__ Wh) {
  const int idx = blockIdx.x * blockDim.x + threadIdx.x;
  if (idx >= 4096) return;
  const int j    = idx & 15;
  const int lane = (idx >> 4) & 31;
  const int cb   = (idx >> 9) & 3;
  const int ks   = idx >> 11;
  const int k    = ks * 32 + (lane >> 4) * 8 + (j & 7) + (j >> 3) * 16;
  const int col  = cb * 16 + (lane & 15);
  Wh[idx] = (_Float16)W[k * 64 + col];
}

// ---------------------------------------------------------------------------
// GEMM: Out[nrows,64] = A[nrows,64] (*rowscale) @ W[64,64] (+bias) (+addend)
// One wave computes a 16x64 strip: 4 C-tiles x 2 K-steps = 8 WMMAs.
// ---------------------------------------------------------------------------
__global__ __launch_bounds__(256)
void gemm64_wmma(const float* __restrict__ A, const _Float16* __restrict__ Wh,
                 const float* __restrict__ bias, const float* __restrict__ addend,
                 const float* __restrict__ rowscale, float* __restrict__ Out,
                 int nrows) {
  const int wave   = threadIdx.x >> 5;
  const int lane   = threadIdx.x & 31;
  const int strip  = blockIdx.x * 8 + wave;
  const int nstrip = (nrows + 15) >> 4;
  if (strip >= nstrip) return;               // uniform per wave -> EXEC all-1s
  const int row0    = strip * 16;
  const int halfsel = lane >> 4;             // 0: lanes 0-15, 1: lanes 16-31
  const int l15     = lane & 15;

  int arow = row0 + l15;
  if (arow >= nrows) arow = nrows - 1;       // clamp reads (N%16==0 normally)
  const float rs = rowscale ? rowscale[arow] : 1.0f;
  const float* arp = A + (size_t)arow * 64;

  v8f acc0 = {}, acc1 = {}, acc2 = {}, acc3 = {};
#pragma unroll
  for (int ks = 0; ks < 2; ++ks) {
    // A fragment: ISA 16-bit A 16x32 layout. Lane reads two contiguous
    // 8-float runs -> four b128 loads, then pack to f16.
    const int kb = ks * 32 + halfsel * 8;
    const float4 f0 = *(const float4*)(arp + kb);
    const float4 f1 = *(const float4*)(arp + kb + 4);
    const float4 f2 = *(const float4*)(arp + kb + 16);
    const float4 f3 = *(const float4*)(arp + kb + 20);
    v16h a;
    a[0] = (_Float16)(f0.x * rs);  a[1] = (_Float16)(f0.y * rs);
    a[2] = (_Float16)(f0.z * rs);  a[3] = (_Float16)(f0.w * rs);
    a[4] = (_Float16)(f1.x * rs);  a[5] = (_Float16)(f1.y * rs);
    a[6] = (_Float16)(f1.z * rs);  a[7] = (_Float16)(f1.w * rs);
    a[8]  = (_Float16)(f2.x * rs); a[9]  = (_Float16)(f2.y * rs);
    a[10] = (_Float16)(f2.z * rs); a[11] = (_Float16)(f2.w * rs);
    a[12] = (_Float16)(f3.x * rs); a[13] = (_Float16)(f3.y * rs);
    a[14] = (_Float16)(f3.z * rs); a[15] = (_Float16)(f3.w * rs);
#pragma unroll
    for (int cb = 0; cb < 4; ++cb) {
      // B fragment: contiguous pre-packed v16h per lane
      const v16h b = *(const v16h*)(Wh + ((((ks * 4 + cb) * 32) + lane) << 4));
      v8f* accp = (cb == 0) ? &acc0 : (cb == 1) ? &acc1 : (cb == 2) ? &acc2 : &acc3;
      *accp = __builtin_amdgcn_wmma_f32_16x16x32_f16(false, a, false, b,
                                                     (short)0, *accp, false, false);
    }
  }
  // store: C/D layout (VGPR r -> M = r + 8*halfsel, N = lane&15)
  const int rbase = row0 + halfsel * 8;
#pragma unroll
  for (int cb = 0; cb < 4; ++cb) {
    const v8f acc = (cb == 0) ? acc0 : (cb == 1) ? acc1 : (cb == 2) ? acc2 : acc3;
    const int col = cb * 16 + l15;
    const float bb = bias ? bias[col] : 0.0f;
#pragma unroll
    for (int r = 0; r < 8; ++r) {
      const int row = rbase + r;
      if (row < nrows) {
        float v = acc[r] + bb;
        if (addend) v += addend[(size_t)row * 64 + col];
        Out[(size_t)row * 64 + col] = v;
      }
    }
  }
}

// ---------------------------------------------------------------------------
// small utility kernels
// ---------------------------------------------------------------------------
__global__ void izero(int* p, int n) {
  int i = blockIdx.x * blockDim.x + threadIdx.x;
  if (i < n) p[i] = 0;
}
__global__ void fzero(float* p, int n) {
  int i = blockIdx.x * blockDim.x + threadIdx.x;
  if (i < n) p[i] = 0.f;
}
__global__ void count_in_degree(const int* __restrict__ dst, int E, int* __restrict__ cnt) {
  int i = blockIdx.x * blockDim.x + threadIdx.x;
  if (i < E) atomicAdd(&cnt[dst[i]], 1);
}
__global__ void make_norms(const int* __restrict__ cnt, int N,
                           float* __restrict__ dinv, float* __restrict__ invcnt) {
  int n = blockIdx.x * blockDim.x + threadIdx.x;
  if (n < N) {
    const float c = (float)cnt[n];
    dinv[n]   = rsqrtf(c + 1.0f);            // GCN degree incl. self loop
    invcnt[n] = 1.0f / fmaxf(c, 1.0f);       // SAGE mean denominator
  }
}
// out = relu(v) + v with v = in (+ bias)
__global__ void bias_relu_res(const float* __restrict__ in, const float* __restrict__ bias,
                              float* __restrict__ out, int NE) {
  int i = blockIdx.x * blockDim.x + threadIdx.x;
  if (i < NE) {
    float v = in[i] + (bias ? bias[i & 63] : 0.f);
    out[i] = fmaxf(v, 0.f) + v;
  }
}

// ---------------------------------------------------------------------------
// GCN aggregation
// ---------------------------------------------------------------------------
__global__ void gcn_init(const float* __restrict__ t, const float* __restrict__ dinv,
                         const float* __restrict__ bias, float* __restrict__ agg, int NE) {
  int i = blockIdx.x * blockDim.x + threadIdx.x;
  if (i < NE) {
    const int n = i >> 6;
    const float di = dinv[n];
    agg[i] = t[i] * di * di + bias[i & 63];  // self loop + bias
  }
}
__global__ void gcn_edges(const float* __restrict__ t, const int* __restrict__ src,
                          const int* __restrict__ dst, const float* __restrict__ dinv,
                          float* __restrict__ agg, int E) {
  int i = blockIdx.x * blockDim.x + threadIdx.x;
  if (i >= E * 16) return;
  const int e = i >> 4, q = i & 15;
  const int s = src[e], d = dst[e];
  const float w = dinv[s] * dinv[d];
  const float4 v = *(const float4*)(t + (size_t)s * 64 + q * 4);
  float* o = agg + (size_t)d * 64 + q * 4;
  atomAddF(o + 0, v.x * w); atomAddF(o + 1, v.y * w);
  atomAddF(o + 2, v.z * w); atomAddF(o + 3, v.w * w);
}

// ---------------------------------------------------------------------------
// SAGE aggregation (sum; mean folded into GEMM row-scale)
// ---------------------------------------------------------------------------
__global__ void sage_edges(const float* __restrict__ x, const int* __restrict__ src,
                           const int* __restrict__ dst, float* __restrict__ agg, int E) {
  int i = blockIdx.x * blockDim.x + threadIdx.x;
  if (i >= E * 16) return;
  const int e = i >> 4, q = i & 15;
  const int s = src[e], d = dst[e];
  const float4 v = *(const float4*)(x + (size_t)s * 64 + q * 4);
  float* o = agg + (size_t)d * 64 + q * 4;
  atomAddF(o + 0, v.x); atomAddF(o + 1, v.y);
  atomAddF(o + 2, v.z); atomAddF(o + 3, v.w);
}

// ---------------------------------------------------------------------------
// GAT: logits, segment max / denom / weighted scatter (self loops in inits)
// ---------------------------------------------------------------------------
__global__ void attn_logits(const float* __restrict__ h1, const float* __restrict__ asrc,
                            const float* __restrict__ adst, float* __restrict__ alS,
                            float* __restrict__ alD, int N, int H) {
  int i = blockIdx.x * blockDim.x + threadIdx.x;
  if (i >= N * H) return;
  const int n = i / H, hd = i % H, C = 64 / H;
  const float* hp = h1 + (size_t)n * 64 + hd * C;
  const float* as = asrc + hd * C;
  const float* ad = adst + hd * C;
  float s = 0.f, d = 0.f;
  for (int c = 0; c < C; ++c) { const float v = hp[c]; s += v * as[c]; d += v * ad[c]; }
  alS[i] = s; alD[i] = d;
}
__global__ void gat_init_max(const float* __restrict__ alS, const float* __restrict__ alD,
                             int* __restrict__ emaxk, int NH) {
  int i = blockIdx.x * blockDim.x + threadIdx.x;
  if (i < NH) emaxk[i] = okey(lrelu(alS[i] + alD[i]));   // self-loop seed
}
__global__ void gat_edge_max(const float* __restrict__ alS, const float* __restrict__ alD,
                             const int* __restrict__ src, const int* __restrict__ dst,
                             int* __restrict__ emaxk, int E, int H) {
  int i = blockIdx.x * blockDim.x + threadIdx.x;
  if (i >= E * H) return;
  const int e = i / H, hd = i % H;
  const int s = src[e], d = dst[e];
  const float v = lrelu(alS[s * H + hd] + alD[d * H + hd]);
  atomicMax(&emaxk[d * H + hd], okey(v));
}
__global__ void gat_den_init(const float* __restrict__ alS, const float* __restrict__ alD,
                             const int* __restrict__ emaxk, float* __restrict__ emaxf,
                             float* __restrict__ den, int NH) {
  int i = blockIdx.x * blockDim.x + threadIdx.x;
  if (i >= NH) return;
  const float m = dekey(emaxk[i]);
  emaxf[i] = m;
  den[i] = expf(lrelu(alS[i] + alD[i]) - m);             // self-loop term
}
__global__ void gat_edge_den(const float* __restrict__ alS, const float* __restrict__ alD,
                             const int* __restrict__ src, const int* __restrict__ dst,
                             const float* __restrict__ emaxf, float* __restrict__ den,
                             int E, int H) {
  int i = blockIdx.x * blockDim.x + threadIdx.x;
  if (i >= E * H) return;
  const int e = i / H, hd = i % H;
  const int s = src[e], d = dst[e];
  const float v = lrelu(alS[s * H + hd] + alD[d * H + hd]);
  atomAddF(&den[d * H + hd], expf(v - emaxf[d * H + hd]));
}
__global__ void gat_out_init(const float* __restrict__ h1, const float* __restrict__ alS,
                             const float* __restrict__ alD, const float* __restrict__ emaxf,
                             const float* __restrict__ den, float* __restrict__ out,
                             int NE, int H) {
  int i = blockIdx.x * blockDim.x + threadIdx.x;
  if (i >= NE) return;
  const int n = i >> 6, c = i & 63, C = 64 / H;
  const int idx = n * H + c / C;
  const float alpha = expf(lrelu(alS[idx] + alD[idx]) - emaxf[idx]) / (den[idx] + 1e-16f);
  out[i] = h1[i] * alpha;                                 // self-loop contribution
}
__global__ void gat_edge_out(const float* __restrict__ h1, const float* __restrict__ alS,
                             const float* __restrict__ alD, const float* __restrict__ emaxf,
                             const float* __restrict__ den, const int* __restrict__ src,
                             const int* __restrict__ dst, float* __restrict__ out,
                             int E, int H) {
  int i = blockIdx.x * blockDim.x + threadIdx.x;
  if (i >= E * 16) return;
  const int e = i >> 4, q = i & 15, C = 64 / H;
  const int s = src[e], d = dst[e];
  const int hd = (q * 4) / C;                             // 4-chunk never crosses a head
  const int idx = d * H + hd;
  const float alpha = expf(lrelu(alS[s * H + hd] + alD[idx]) - emaxf[idx]) / (den[idx] + 1e-16f);
  const float4 v = *(const float4*)(h1 + (size_t)s * 64 + q * 4);
  float* o = out + (size_t)d * 64 + q * 4;
  atomAddF(o + 0, v.x * alpha); atomAddF(o + 1, v.y * alpha);
  atomAddF(o + 2, v.z * alpha); atomAddF(o + 3, v.w * alpha);
}

// ---------------------------------------------------------------------------
// in-place LayerNorm over 64-wide rows, one wave per row (wave32 shuffles)
// ---------------------------------------------------------------------------
__global__ __launch_bounds__(256)
void layernorm64(float* __restrict__ data, const float* __restrict__ g,
                 const float* __restrict__ b, int rows) {
  const int row = blockIdx.x * 8 + (threadIdx.x >> 5);
  if (row >= rows) return;
  const int lane = threadIdx.x & 31;
  float* p = data + (size_t)row * 64;
  const float x0 = p[lane * 2], x1 = p[lane * 2 + 1];
  float s = x0 + x1, ss = x0 * x0 + x1 * x1;
#pragma unroll
  for (int m = 16; m >= 1; m >>= 1) {
    s  += __shfl_xor(s, m, 32);
    ss += __shfl_xor(ss, m, 32);
  }
  const float mu  = s * (1.0f / 64.0f);
  const float var = ss * (1.0f / 64.0f) - mu * mu;
  const float inv = rsqrtf(var + 1e-6f);
  p[lane * 2]     = (x0 - mu) * inv * g[lane * 2]     + b[lane * 2];
  p[lane * 2 + 1] = (x1 - mu) * inv * g[lane * 2 + 1] + b[lane * 2 + 1];
}
__global__ void write_batchs(const int* __restrict__ batch, float* __restrict__ out, int N) {
  int i = blockIdx.x * blockDim.x + threadIdx.x;
  if (i < 3 * N) out[i] = (float)batch[i % N];
}

// ---------------------------------------------------------------------------
// orchestration
// ---------------------------------------------------------------------------
static inline int NB(long n) { return (int)((n + 255) / 256); }

extern "C" void kernel_launch(void* const* d_in, const int* in_sizes, int n_in,
                              void* d_out, int out_size, void* d_ws, size_t ws_size,
                              hipStream_t stream) {
  const int N = in_sizes[0] / 64;
  const int E = in_sizes[1] / 2;

  const float* x       = (const float*)d_in[0];
  const int*   ei      = (const int*)d_in[1];
  const int*   src     = ei;
  const int*   dst     = ei + E;
  const int*   batch   = (const int*)d_in[2];
  const float* gcn_W   = (const float*)d_in[3];
  const float* gcn_b   = (const float*)d_in[4];
  const float* sage_Wl = (const float*)d_in[5];
  const float* sage_bl = (const float*)d_in[6];
  const float* sage_Wr = (const float*)d_in[7];
  const float* gat_W[2]  = { (const float*)d_in[8],  (const float*)d_in[12] };
  const float* gat_as[2] = { (const float*)d_in[9],  (const float*)d_in[13] };
  const float* gat_ad[2] = { (const float*)d_in[10], (const float*)d_in[14] };
  const float* gat_b[2]  = { (const float*)d_in[11], (const float*)d_in[15] };
  const float* ln_g = (const float*)d_in[16];
  const float* ln_b = (const float*)d_in[17];
  float* out = (float*)d_out;

  // workspace layout: Wh16 first (256B-aligned base => aligned v16h loads)
  // slots: 0-2 gcn_W[l], 3 gat1_W, 4 gat2_W, 5-7 sage_Wl[l], 8-10 sage_Wr[l]
  _Float16* wh   = (_Float16*)d_ws;            // 11 * 4096 halfs (90KB)
  float* w       = (float*)(wh + 11 * 4096);
  float* dinv    = w;                          // N
  float* invcnt  = dinv + N;                   // N
  float* hbuf    = invcnt + N;                 // N*64
  float* t       = hbuf + (size_t)N * 64;      // N*64
  float* agg     = t    + (size_t)N * 64;      // N*64
  float* alS     = agg  + (size_t)N * 64;      // N*8
  float* alD     = alS + (size_t)N * 8;        // N*8
  float* emaxf   = alD + (size_t)N * 8;        // N*8
  float* den     = emaxf + (size_t)N * 8;      // N*8
  int*   emaxk   = (int*)(den + (size_t)N * 8);// N*8
  int*   cnt     = emaxk + (size_t)N * 8;      // N

  const dim3 blk(256);
  const int NE = N * 64;
  const int gemm_blocks = (((N + 15) / 16) + 7) / 8;

  // ---- pre-pack all 11 weight matrices to f16 fragment layout ----
  const float* wsrc[11] = { gcn_W, gcn_W + 4096, gcn_W + 8192,
                            gat_W[0], gat_W[1],
                            sage_Wl, sage_Wl + 4096, sage_Wl + 8192,
                            sage_Wr, sage_Wr + 4096, sage_Wr + 8192 };
  for (int m = 0; m < 11; ++m)
    pack_w16<<<16, blk, 0, stream>>>(wsrc[m], wh + (size_t)m * 4096);
  _Float16* whv[11];
  for (int m = 0; m < 11; ++m) whv[m] = wh + (size_t)m * 4096;

  // ---- degrees / norms (shared by GCN + SAGE) ----
  izero<<<NB(N), blk, 0, stream>>>(cnt, N);
  count_in_degree<<<NB(E), blk, 0, stream>>>(dst, E, cnt);
  make_norms<<<NB(N), blk, 0, stream>>>(cnt, N, dinv, invcnt);

  // ---- channel 0: 3x GCN -> out[0:N*64] ----
  {
    const float* cur = x;
    for (int l = 0; l < 3; ++l) {
      gemm64_wmma<<<gemm_blocks, blk, 0, stream>>>(cur, whv[l],
                                                   nullptr, nullptr, nullptr, t, N);
      gcn_init<<<NB(NE), blk, 0, stream>>>(t, dinv, gcn_b + l * 64, agg, NE);
      gcn_edges<<<NB((long)E * 16), blk, 0, stream>>>(t, src, dst, dinv, agg, E);
      float* op = (l == 2) ? out : hbuf;
      bias_relu_res<<<NB(NE), blk, 0, stream>>>(agg, nullptr, op, NE);
      cur = op;
    }
  }

  // ---- channel 1: GAT(8 heads) -> GAT(1 head) -> out[N*64:2*N*64] ----
  for (int layer = 0; layer < 2; ++layer) {
    const int H = (layer == 0) ? 8 : 1;
    const float* inp = (layer == 0) ? x : hbuf;
    float* op = (layer == 0) ? hbuf : out + (size_t)N * 64;
    gemm64_wmma<<<gemm_blocks, blk, 0, stream>>>(inp, whv[3 + layer],
                                                 nullptr, nullptr, nullptr, t, N);
    attn_logits<<<NB((long)N * H), blk, 0, stream>>>(t, gat_as[layer], gat_ad[layer],
                                                     alS, alD, N, H);
    gat_init_max<<<NB((long)N * H), blk, 0, stream>>>(alS, alD, emaxk, N * H);
    gat_edge_max<<<NB((long)E * H), blk, 0, stream>>>(alS, alD, src, dst, emaxk, E, H);
    gat_den_init<<<NB((long)N * H), blk, 0, stream>>>(alS, alD, emaxk, emaxf, den, N * H);
    gat_edge_den<<<NB((long)E * H), blk, 0, stream>>>(alS, alD, src, dst, emaxf, den, E, H);
    gat_out_init<<<NB(NE), blk, 0, stream>>>(t, alS, alD, emaxf, den, agg, NE, H);
    gat_edge_out<<<NB((long)E * 16), blk, 0, stream>>>(t, alS, alD, emaxf, den,
                                                       src, dst, agg, E, H);
    bias_relu_res<<<NB(NE), blk, 0, stream>>>(agg, gat_b[layer], op, NE);
  }

  // ---- channel 2: 3x SAGE -> out[2*N*64:3*N*64] ----
  {
    const float* cur = x;
    for (int l = 0; l < 3; ++l) {
      // t = cur @ Wr  (root transform)
      gemm64_wmma<<<gemm_blocks, blk, 0, stream>>>(cur, whv[8 + l],
                                                   nullptr, nullptr, nullptr, t, N);
      // agg = segment_sum(cur[src] -> dst)
      fzero<<<NB(NE), blk, 0, stream>>>(agg, NE);
      sage_edges<<<NB((long)E * 16), blk, 0, stream>>>(cur, src, dst, agg, E);
      // t = (agg * invcnt) @ Wl + bl + t    (mean folded into row-scale; in-place addend)
      gemm64_wmma<<<gemm_blocks, blk, 0, stream>>>(agg, whv[5 + l],
                                                   sage_bl + l * 64, t, invcnt, t, N);
      float* op = (l == 2) ? out + (size_t)2 * N * 64 : hbuf;
      bias_relu_res<<<NB(NE), blk, 0, stream>>>(t, nullptr, op, NE);
      cur = op;
    }
  }

  // ---- LayerNorm over [3,N,64] in-place, then batchs ----
  layernorm64<<<(3 * N + 7) / 8, blk, 0, stream>>>(out, ln_g, ln_b, 3 * N);
  write_batchs<<<NB((long)3 * N), blk, 0, stream>>>(batch, out + (size_t)3 * N * 64, N);
}
